// GeometryAwareTensorAttention_90426241450251
// MI455X (gfx1250) — matrix operand order repaired and checked by reading, not compiled, not run on hardware
//
#include <hip/hip_runtime.h>

typedef __attribute__((ext_vector_type(16))) _Float16 v16h;
typedef __attribute__((ext_vector_type(8)))  float    v8f;

#define BATCH 2
#define NPTS  1024
#define MNB   32
#define DIMC  128
#define HEADS 8
#define DH    16
#define SDIM  5
#define DMLP  256
#define NB    (BATCH*NPTS)          // 2048 rows
#define SDI   (SDIM*DIMC)           // 640

// ---------------------------------------------------------------------------
// WMMA fragment layouts (CDNA5 ISA 7.12.2, wave32), stored "fragment-packed":
// each 16x32 (A) or 32x16 (B) tile = 32 lanes x 16 f16, lane-contiguous 32 B.
// A element (m,k):  lane = (m&15) | (((k>>3)&1)<<4), slot = ((k>>4)<<3)|(k&7)
// B element (k,n):  lane = (n&15) | (((k>>3)&1)<<4), slot = ((k>>4)<<3)|(k&7)
// C/D 16x16 f32:    lane l -> col (l&15); VGPR v -> row v + (l>>4)*8
// ---------------------------------------------------------------------------
__device__ inline v16h frag_load(const _Float16* P, int fragIdx, int lane) {
    const uint4* p = (const uint4*)(P + ((size_t)fragIdx * 32 + lane) * 16);
    union { uint4 u[2]; v16h v; } un;
    un.u[0] = p[0];
    un.u[1] = p[1];
    return un.v;
}

__device__ inline size_t a_pos(int frag, int m, int kk) {   // kk = k & 31
    int lane = (m & 15) | (((kk >> 3) & 1) << 4);
    int slot = ((kk >> 4) << 3) | (kk & 7);
    return ((size_t)frag * 32 + lane) * 16 + slot;
}

// RA/RB = operand-reuse hints (VOP3P OPSEL/OPSEL_HI), set on the *subsequent*
// of two identical consecutive WMMAs that share the A (RA) or B (RB) operand.
template <bool RA, bool RB>
__device__ inline v8f wmma_f16(v16h a, v16h b, v8f c) {
    return __builtin_amdgcn_wmma_f32_16x16x32_f16(false, a, false, b, (short)0, c, RA, RB);
}

__device__ inline float rcp_fast(float x) { return __builtin_amdgcn_rcpf(x); }
__device__ inline float silu_f(float x)   { return x * rcp_fast(1.f + __expf(-x)); }
__device__ inline float sigmoid_f(float x){ return rcp_fast(1.f + __expf(-x)); }

// ---------------------------------------------------------------------------
// Weight packer: f32 (KxN row-major) -> f16 B-fragment-packed.
// ---------------------------------------------------------------------------
__global__ void pack_b_kernel(const float* __restrict__ W, _Float16* __restrict__ P,
                              int K, int N) {
    int i = blockIdx.x * blockDim.x + threadIdx.x;
    if (i >= K * N) return;
    int k = i / N, n = i % N;
    int kk = k & 31;
    int frag = (n >> 4) * (K >> 5) + (k >> 5);
    int lane = (n & 15) | (((kk >> 3) & 1) << 4);
    int slot = ((kk >> 4) << 3) | (kk & 7);
    P[((size_t)frag * 32 + lane) * 16 + slot] = (_Float16)W[i];
}

// ---------------------------------------------------------------------------
// LayerNorm: 128 threads per row; hi f32 (for gates) + hi/hj f16 A-packed.
// ---------------------------------------------------------------------------
__global__ __launch_bounds__(128) void ln_kernel(const float* __restrict__ h,
                                                 const float* __restrict__ wi,
                                                 const float* __restrict__ wj,
                                                 float* __restrict__ hiF,
                                                 _Float16* __restrict__ hiP,
                                                 _Float16* __restrict__ hjP) {
    int row = blockIdx.x;
    int d = threadIdx.x;
    __shared__ float red[128];
    float x = h[row * DIMC + d];
    red[d] = x; __syncthreads();
    for (int off = 64; off > 0; off >>= 1) { if (d < off) red[d] += red[d + off]; __syncthreads(); }
    float mu = red[0] * (1.f / DIMC);
    __syncthreads();
    float xm = x - mu;
    red[d] = xm * xm; __syncthreads();
    for (int off = 64; off > 0; off >>= 1) { if (d < off) red[d] += red[d + off]; __syncthreads(); }
    float inv = rsqrtf(red[0] * (1.f / DIMC) + 1e-5f);
    float xn = xm * inv;
    float hi = xn * wi[d];
    float hj = xn * wj[d];
    hiF[row * DIMC + d] = hi;
    size_t pos = a_pos((row >> 4) * (DIMC >> 5) + (d >> 5), row, d & 31);
    hiP[pos] = (_Float16)hi;
    hjP[pos] = (_Float16)hj;
}

// ---------------------------------------------------------------------------
// Single-wave WMMA GEMM, 16x64 per wave: A fragments loaded once into
// registers and amortized over 4 column tiles (reuse_a on the shared-A chain).
// K is a template param so all fragment arrays stay in registers.
// ACT: 0 none, 1 silu.  OUTM: 0 -> f32 row-major, 1 -> f16 A-packed.
// ---------------------------------------------------------------------------
template <int K, int ACT, int OUTM>
__global__ __launch_bounds__(32) void gemm_wmma(const _Float16* __restrict__ A,
                                                const _Float16* __restrict__ W,
                                                const float* __restrict__ bias,
                                                float* __restrict__ outF,
                                                _Float16* __restrict__ outP,
                                                int Ncols) {
    constexpr int KT = K >> 5;
    int mr  = blockIdx.x * 16;
    int nb0 = blockIdx.y * 64;
    int lane = threadIdx.x & 31;

    v16h af[KT];
#pragma unroll
    for (int kb = 0; kb < KT; ++kb) af[kb] = frag_load(A, (mr >> 4) * KT + kb, lane);

    v8f c[4] = {};
#pragma unroll
    for (int kb = 0; kb < KT; ++kb) {
        v16h bf[4];
#pragma unroll
        for (int nt = 0; nt < 4; ++nt)
            bf[nt] = frag_load(W, ((nb0 >> 4) + nt) * KT + kb, lane);
        c[0] = wmma_f16<false, false>(af[kb], bf[0], c[0]);
        c[1] = wmma_f16<true,  false>(af[kb], bf[1], c[1]);
        c[2] = wmma_f16<true,  false>(af[kb], bf[2], c[2]);
        c[3] = wmma_f16<true,  false>(af[kb], bf[3], c[3]);
    }

#pragma unroll
    for (int nt = 0; nt < 4; ++nt) {
        int n = nb0 + nt * 16 + (lane & 15);
        float bv = bias ? bias[n] : 0.f;
#pragma unroll
        for (int vi = 0; vi < 8; ++vi) {
            int m = mr + vi + ((lane >> 4) << 3);
            float val = c[nt][vi] + bv;
            if (ACT == 1) val = silu_f(val);
            if (OUTM == 0) {
                outF[(size_t)m * Ncols + n] = val;
            } else {
                outP[a_pos((m >> 4) * (Ncols >> 5) + (n >> 5), m, n & 31)] = (_Float16)val;
            }
        }
    }
}

// gates = sigmoid(hi @ Wg + bg): 128x40 per row, tiny -> VALU.
__global__ __launch_bounds__(64) void gates_kernel(const float* __restrict__ hiF,
                                                   const float* __restrict__ Wg,
                                                   const float* __restrict__ bg,
                                                   float* __restrict__ gates) {
    int row = blockIdx.x;
    int t = threadIdx.x;
    if (t < HEADS * SDIM) {
        float acc = bg[t];
        for (int d = 0; d < DIMC; ++d) acc += hiF[row * DIMC + d] * Wg[d * (HEADS * SDIM) + t];
        gates[row * (HEADS * SDIM) + t] = sigmoid_f(acc);
    }
}

// ---------------------------------------------------------------------------
// Fused per-node kernel (8 waves). The full 32x128 t_ij A-operand (8
// fragments, 64 VGPRs) is loaded ONCE into registers and reused across all
// Wek column tiles (phase 1) and all Wev tiles (phase 3a). The two row-tiles
// share each B fragment -> reuse_b on the second WMMA of every pair.
// ---------------------------------------------------------------------------
__global__ __launch_bounds__(256) void attn_kernel(
    const float* __restrict__ t_ij, const float* __restrict__ r1,
    const float* __restrict__ r2,   const float* __restrict__ x1,
    const float* __restrict__ x2,   const int* __restrict__ nidx,
    const unsigned char* __restrict__ nmask,
    const float* __restrict__ qG,   const float* __restrict__ kG,
    const float* __restrict__ vG,   const float* __restrict__ pvG,
    const float* __restrict__ gatesG,
    const _Float16* __restrict__ WekP, const _Float16* __restrict__ WevP,
    const _Float16* __restrict__ WoP,
    float* __restrict__ h_res, float* __restrict__ xres1, float* __restrict__ xres2) {

    const int node = blockIdx.x;           // b*N + i
    const int tid  = threadIdx.x;
    const int lane = tid & 31;
    const int w    = tid >> 5;
    const int brow = (node >> 10) << 10;   // b*N

    __shared__ __align__(32) _Float16 t_pack[8 * 32 * 16];  // 8 KB, A-packed t_ij
    __shared__ __align__(32) _Float16 x_pack[8 * 32 * 16];  // 8 KB, A-packed X
    __shared__ _Float16 kn_h[MNB][DIMC];   // 8 KB  gathered k rows (f16)
    __shared__ float    q_s[DIMC];
    __shared__ float    gates_s[HEADS * SDIM];
    __shared__ int      idx_s[MNB];
    __shared__ float    msk_s[MNB];
    __shared__ float    sim_s[HEADS][MNB][SDIM];  // 5 KB  sim -> attn
    __shared__ float    ev_s[MNB][DIMC];   // 16 KB ev tile for current s
    __shared__ float    out_s[MNB][DIMC];  // 16 KB out tile for current s

    if (tid < MNB) {
        idx_s[tid] = nidx[node * MNB + tid];
        msk_s[tid] = nmask[node * MNB + tid] ? 1.f : 0.f;
    }
    if (tid < DIMC) q_s[tid] = qG[node * DIMC + tid];
    if (tid >= 128 && tid < 128 + HEADS * SDIM)
        gates_s[tid - 128] = gatesG[node * (HEADS * SDIM) + (tid - 128)];
    __syncthreads();

    for (int e = tid; e < MNB * DIMC; e += 256) {
        int m = e >> 7, d = e & 127;
        _Float16 tv = (_Float16)t_ij[(size_t)(node * MNB + m) * DIMC + d];
        t_pack[a_pos((m >> 4) * 4 + (d >> 5), m, d & 31)] = tv;
        kn_h[m][d] = (_Float16)kG[(size_t)(brow + idx_s[m]) * DIMC + d];
    }
    __syncthreads();

    // ---- hoist the full t_ij A operand into registers (8 frags, 64 VGPR) --
    v16h ta0[4], ta1[4];
#pragma unroll
    for (int kb = 0; kb < 4; ++kb) {
        ta0[kb] = frag_load(t_pack, kb, lane);       // rows 0..15
        ta1[kb] = frag_load(t_pack, 4 + kb, lane);   // rows 16..31
    }

    // -------- Phase 1: sim[h][m][s] = sum_dh q*k_n*silu(t_ij@Wek) ----------
    for (int ci = 0; ci < 5; ++ci) {
        int ct = w + ci * 8;               // column tile 0..39 of Wek
        int s = ct >> 3, hh = ct & 7;
        v16h bf[4];
#pragma unroll
        for (int kb = 0; kb < 4; ++kb) bf[kb] = frag_load(WekP, ct * 4 + kb, lane);
        v8f c0 = {}, c1 = {};
#pragma unroll
        for (int kb = 0; kb < 4; ++kb) {
            c0 = wmma_f16<false, false>(ta0[kb], bf[kb], c0);
            c1 = wmma_f16<false, true >(ta1[kb], bf[kb], c1);   // shares B
        }
        int dh = lane & 15;
        float qv = q_s[hh * DH + dh];
#pragma unroll
        for (int vi = 0; vi < 8; ++vi) {
            int m0 = vi + ((lane >> 4) << 3);
            float p0 = qv * (float)kn_h[m0][hh * DH + dh] * silu_f(c0[vi]);
            float p1 = qv * (float)kn_h[m0 + 16][hh * DH + dh] * silu_f(c1[vi]);
            p0 += __shfl_xor(p0, 1);  p1 += __shfl_xor(p1, 1);
            p0 += __shfl_xor(p0, 2);  p1 += __shfl_xor(p1, 2);
            p0 += __shfl_xor(p0, 4);  p1 += __shfl_xor(p1, 4);
            p0 += __shfl_xor(p0, 8);  p1 += __shfl_xor(p1, 8);
            if ((lane & 15) == 0) {
                sim_s[hh][m0][s]      = p0;
                sim_s[hh][m0 + 16][s] = p1;
            }
        }
    }
    __syncthreads();

    // -------- Phase 2: tanh clamp + masked softmax over M ------------------
    if (tid < HEADS * SDIM) {
        int hh = tid / SDIM, s = tid % SDIM;
        const float NEG = -3.402823466e38f;
        float vals[MNB];
        float mx = NEG;
        for (int m = 0; m < MNB; ++m) {
            float sv = sim_s[hh][m][s];
            sv = tanhf(sv * (1.f / 50.f)) * 50.f;
            if (msk_s[m] == 0.f) sv = NEG;
            vals[m] = sv;
            mx = fmaxf(mx, sv);
        }
        float sum = 0.f;
        for (int m = 0; m < MNB; ++m) { vals[m] = __expf(vals[m] - mx); sum += vals[m]; }
        float rs = rcp_fast(sum);
        for (int m = 0; m < MNB; ++m) sim_s[hh][m][s] = vals[m] * rs;
    }
    __syncthreads();

    // -------- Phase 3: per-s ev tile, gated X, X @ Wo, reductions ----------
    for (int s = 0; s < SDIM; ++s) {
        // 3a: ev = t_ij @ Wev (t-fragments already in registers)
        {
            int hh = w;                    // wave w owns head-tile w
            int ct = s * 8 + hh;
            v16h bf[4];
#pragma unroll
            for (int kb = 0; kb < 4; ++kb) bf[kb] = frag_load(WevP, ct * 4 + kb, lane);
            v8f c0 = {}, c1 = {};
#pragma unroll
            for (int kb = 0; kb < 4; ++kb) {
                c0 = wmma_f16<false, false>(ta0[kb], bf[kb], c0);
                c1 = wmma_f16<false, true >(ta1[kb], bf[kb], c1);
            }
            int n = lane & 15;
#pragma unroll
            for (int vi = 0; vi < 8; ++vi) {
                int m = vi + ((lane >> 4) << 3);
                ev_s[m][hh * DH + n]      = c0[vi];
                ev_s[m + 16][hh * DH + n] = c1[vi];
            }
        }
        __syncthreads();

        // 3b: X[m,di] = (attn*v_n + ev*pv_n) * gate  -> A-packed LDS
        for (int e = tid; e < MNB * DIMC; e += 256) {
            int m = e >> 7, di = e & 127;
            int hh = di >> 4;
            int col = s * DIMC + di;
            size_t row = (size_t)(brow + idx_s[m]);
            float at = sim_s[hh][m][s];
            float xval = (at * vG[row * SDI + col] + ev_s[m][di] * pvG[row * SDI + col]) *
                         gates_s[hh * SDIM + s];
            x_pack[a_pos((m >> 4) * 4 + (di >> 5), m, di & 31)] = (_Float16)xval;
        }
        __syncthreads();

        // 3c: out = X @ Wo
        {
            int hh = w;                    // col tile of Wo
            v16h xa0[4], xa1[4];
#pragma unroll
            for (int kb = 0; kb < 4; ++kb) {
                xa0[kb] = frag_load(x_pack, kb, lane);
                xa1[kb] = frag_load(x_pack, 4 + kb, lane);
            }
            v16h bf[4];
#pragma unroll
            for (int kb = 0; kb < 4; ++kb) bf[kb] = frag_load(WoP, hh * 4 + kb, lane);
            v8f c0 = {}, c1 = {};
#pragma unroll
            for (int kb = 0; kb < 4; ++kb) {
                c0 = wmma_f16<false, false>(xa0[kb], bf[kb], c0);
                c1 = wmma_f16<false, true >(xa1[kb], bf[kb], c1);
            }
            int n = lane & 15;
#pragma unroll
            for (int vi = 0; vi < 8; ++vi) {
                int m = vi + ((lane >> 4) << 3);
                out_s[m][hh * DH + n]      = c0[vi];
                out_s[m + 16][hh * DH + n] = c1[vi];
            }
        }
        __syncthreads();

        // 3d: fused reductions to global outputs
        if (s == 0) {
            for (int d = tid; d < DIMC; d += 256) {
                float acc = 0.f;
                for (int m = 0; m < MNB; ++m) acc += out_s[m][d];
                h_res[(size_t)node * DIMC + d] = acc;
            }
        } else if (s == 1 || s == 2) {
            int comps = (s == 1) ? 3 : 5;
            const float* r = (s == 1) ? r1 : r2;
            float* xr = (s == 1) ? xres1 : xres2;
            for (int e = tid; e < DIMC * comps; e += 256) {
                int d = e / comps, mu = e % comps;
                float acc = 0.f;
                for (int m = 0; m < MNB; ++m)
                    acc += r[(size_t)(node * MNB + m) * comps + mu] * out_s[m][d];
                xr[((size_t)node * DIMC + d) * comps + mu] = acc;   // first writer
            }
        } else {
            int comps = (s == 3) ? 3 : 5;
            const float* xg = (s == 3) ? x1 : x2;
            float* xr = (s == 3) ? xres1 : xres2;
            for (int e = tid; e < DIMC * comps; e += 256) {
                int d = e / comps, mu = e % comps;
                float acc = 0.f;
                for (int m = 0; m < MNB; ++m)
                    acc += xg[((size_t)(brow + idx_s[m]) * DIMC + d) * comps + mu] * out_s[m][d];
                xr[((size_t)node * DIMC + d) * comps + mu] += acc;  // same-block accum
            }
        }
        __syncthreads();
    }
}

// ---------------------------------------------------------------------------
extern "C" void kernel_launch(void* const* d_in, const int* in_sizes, int n_in,
                              void* d_out, int out_size, void* d_ws, size_t ws_size,
                              hipStream_t stream) {
    const float* h      = (const float*)d_in[0];
    const float* t_ij   = (const float*)d_in[1];
    const float* r1     = (const float*)d_in[2];
    const float* r2     = (const float*)d_in[3];
    const float* x1     = (const float*)d_in[4];
    const float* x2     = (const float*)d_in[5];
    const int*   nidx   = (const int*)d_in[6];
    const unsigned char* nmask = (const unsigned char*)d_in[7];
    const float* ln_i_w = (const float*)d_in[8];
    const float* ln_j_w = (const float*)d_in[9];
    const float* Wq     = (const float*)d_in[10];
    const float* Wk     = (const float*)d_in[11];
    const float* Wv1    = (const float*)d_in[12];
    const float* bv1    = (const float*)d_in[13];
    const float* Wv2    = (const float*)d_in[14];
    const float* bv2    = (const float*)d_in[15];
    const float* Wpv1   = (const float*)d_in[16];
    const float* bpv1   = (const float*)d_in[17];
    const float* Wpv2   = (const float*)d_in[18];
    const float* bpv2   = (const float*)d_in[19];
    const float* Wek    = (const float*)d_in[20];
    const float* Wev    = (const float*)d_in[21];
    const float* Wg     = (const float*)d_in[22];
    const float* bg     = (const float*)d_in[23];
    const float* Wo     = (const float*)d_in[24];

    size_t off = 0;
    auto carve = [&](size_t bytes) -> void* {
        void* p = (char*)d_ws + off;
        off += (bytes + 255) & ~(size_t)255;
        return p;
    };
    float*    hiF   = (float*)   carve((size_t)NB * DIMC * 4);
    _Float16* hiP   = (_Float16*)carve((size_t)NB * DIMC * 2);
    _Float16* hjP   = (_Float16*)carve((size_t)NB * DIMC * 2);
    _Float16* WqP   = (_Float16*)carve((size_t)DIMC * DIMC * 2);
    _Float16* WkP   = (_Float16*)carve((size_t)DIMC * DIMC * 2);
    _Float16* Wv1P  = (_Float16*)carve((size_t)DIMC * DMLP * 2);
    _Float16* Wv2P  = (_Float16*)carve((size_t)DMLP * SDI * 2);
    _Float16* Wpv1P = (_Float16*)carve((size_t)DIMC * DMLP * 2);
    _Float16* Wpv2P = (_Float16*)carve((size_t)DMLP * SDI * 2);
    _Float16* WekP  = (_Float16*)carve((size_t)DIMC * SDI * 2);
    _Float16* WevP  = (_Float16*)carve((size_t)DIMC * SDI * 2);
    _Float16* WoP   = (_Float16*)carve((size_t)DIMC * DIMC * 2);
    float*    qF    = (float*)   carve((size_t)NB * DIMC * 4);
    float*    kF    = (float*)   carve((size_t)NB * DIMC * 4);
    _Float16* uvP   = (_Float16*)carve((size_t)NB * DMLP * 2);
    _Float16* upvP  = (_Float16*)carve((size_t)NB * DMLP * 2);
    float*    vF    = (float*)   carve((size_t)NB * SDI * 4);
    float*    pvF   = (float*)   carve((size_t)NB * SDI * 4);
    float*    gatesF= (float*)   carve((size_t)NB * (HEADS * SDIM) * 4);

    float* outp  = (float*)d_out;
    float* h_res = outp;                                    // B*N*128
    float* xres1 = outp + (size_t)NB * DIMC;                // B*N*128*3
    float* xres2 = xres1 + (size_t)NB * DIMC * 3;           // B*N*128*5

    // 1) LayerNorm (writes packed f16 hi/hj)
    ln_kernel<<<NB, 128, 0, stream>>>(h, ln_i_w, ln_j_w, hiF, hiP, hjP);

    // 2) pack weights into B-fragment order (one-time, tiny)
    struct { const float* s; _Float16* d; int K; int N; } cv[9] = {
        {Wq,  WqP,  DIMC, DIMC}, {Wk,  WkP,  DIMC, DIMC},
        {Wv1, Wv1P, DIMC, DMLP}, {Wv2, Wv2P, DMLP, SDI},
        {Wpv1,Wpv1P,DIMC, DMLP}, {Wpv2,Wpv2P,DMLP, SDI},
        {Wek, WekP, DIMC, SDI},  {Wev, WevP, DIMC, SDI},
        {Wo,  WoP,  DIMC, DIMC}};
    for (int i = 0; i < 9; ++i) {
        int n = cv[i].K * cv[i].N;
        pack_b_kernel<<<(n + 255) / 256, 256, 0, stream>>>(cv[i].s, cv[i].d, cv[i].K, cv[i].N);
    }

    // 3) dense projections (WMMA GEMMs, 16x64 per wave)
    gemm_wmma<DIMC,0,0><<<dim3(NB/16, DIMC/64), 32, 0, stream>>>(hiP, WqP,  nullptr, qF, nullptr, DIMC);
    gemm_wmma<DIMC,0,0><<<dim3(NB/16, DIMC/64), 32, 0, stream>>>(hjP, WkP,  nullptr, kF, nullptr, DIMC);
    gemm_wmma<DIMC,1,1><<<dim3(NB/16, DMLP/64), 32, 0, stream>>>(hjP, Wv1P, bv1,  nullptr, uvP,  DMLP);
    gemm_wmma<DIMC,1,1><<<dim3(NB/16, DMLP/64), 32, 0, stream>>>(hjP, Wpv1P,bpv1, nullptr, upvP, DMLP);
    gemm_wmma<DMLP,0,0><<<dim3(NB/16, SDI/64),  32, 0, stream>>>(uvP, Wv2P, bv2,  vF,  nullptr, SDI);
    gemm_wmma<DMLP,0,0><<<dim3(NB/16, SDI/64),  32, 0, stream>>>(upvP,Wpv2P,bpv2, pvF, nullptr, SDI);

    // 4) gates (tiny VALU)
    gates_kernel<<<NB, 64, 0, stream>>>(hiF, Wg, bg, gatesF);

    // 5) fused per-node attention + output projection + residual reductions
    attn_kernel<<<NB, 256, 0, stream>>>(t_ij, r1, r2, x1, x2, nidx, nmask,
                                        qF, kF, vF, pvF, gatesF,
                                        WekP, WevP, WoP,
                                        h_res, xres1, xres2);
    (void)in_sizes; (void)n_in; (void)out_size; (void)ws_size;
}
